// GPTMini_73186242724009
// MI455X (gfx1250) — matrix operand order repaired and checked by reading, not compiled, not run on hardware
//
#include <hip/hip_runtime.h>

// ---------------- problem constants ----------------
#define NB   8
#define SEQ  2048
#define DM   1024
#define HD   32
#define TOK  (NB * SEQ)              // 16384 rows
#define KSTEPS (DM / 32)             // 32 wmma K-steps

typedef __attribute__((ext_vector_type(16))) __bf16 v16bf;
typedef __attribute__((ext_vector_type(8)))  float  v8f;

union Frag32 { uint4 u[2]; v16bf v; };
static_assert(sizeof(Frag32) == 32, "frag size");

__device__ __forceinline__ unsigned short f2bf(float f) {
  unsigned int u = __float_as_uint(f);
  unsigned int r = u + 0x7FFFu + ((u >> 16) & 1u);   // round-to-nearest-even
  return (unsigned short)(r >> 16);
}
__device__ __forceinline__ float bf2f(unsigned short h) {
  return __uint_as_float(((unsigned int)h) << 16);
}

// ---------------- phase 1: h = emb[x] + sin(pos) -> bf16 ----------------
__global__ __launch_bounds__(256) void gpt_embed_pos(
    const int* __restrict__ x, const float* __restrict__ emb,
    unsigned short* __restrict__ h) {
  const int token = blockIdx.x;                 // 0..16383
  const int s     = token & (SEQ - 1);
  const int row   = x[token];
  const float* er = emb + (size_t)row * DM;
  const float sf  = (float)s;
  const int d0 = threadIdx.x * 4;
#pragma unroll
  for (int i = 0; i < 4; ++i) {
    int d = d0 + i;
    // invfreq = 10000^(-2d/D) = exp(-d * 2*ln(10000)/D)
    float invf = __expf((float)d * -0.0179889460745f);
    float p = __sinf(sf * invf);
    h[(size_t)token * DM + d] = f2bf(er[d] + p);
  }
}

// ---------------- phase 2: pack W (N,K) into WMMA B-fragment order ----------
// layout: wp[((ntg*KSTEPS + kt)*32 + lane)*16 + e]  with
//   n = ntg*16 + (lane&15),  k = kt*32 + (lane>>4)*16 + e
__global__ __launch_bounds__(256) void gpt_pack_w(
    const float* __restrict__ W, unsigned short* __restrict__ wp) {
  int idx  = blockIdx.x * 256 + threadIdx.x;    // 0 .. DM*DM-1
  int e    = idx & 15;
  int lane = (idx >> 4) & 31;
  int kt   = (idx >> 9) & (KSTEPS - 1);
  int ntg  = idx >> 14;                          // 0..63
  int n = ntg * 16 + (lane & 15);
  int k = kt * 32 + (lane >> 4) * 16 + e;
  wp[idx] = f2bf(W[(size_t)n * DM + k]);         // B[k][n] = W[n][k]
}

// ---------------- phase 3: q/k/v = h @ W.T + b  (bf16 WMMA) ----------------
// block = 256 thr = 8 waves; block tile 128(M) x 64(N); wave tile 32x32.
__global__ __launch_bounds__(256) void gpt_gemm_qkv(
    const unsigned short* __restrict__ h,     // [TOK][DM] bf16
    const unsigned short* __restrict__ wp,    // packed B fragments
    const float* __restrict__ bias,           // [DM]
    unsigned short* __restrict__ out) {       // [TOK][DM] bf16
  const int lane = threadIdx.x & 31;
  const int w    = threadIdx.x >> 5;
  const int wm = w & 3, wn = w >> 2;                  // 4x2 wave grid
  const int rowbase = blockIdx.x * 128 + wm * 32;
  const int colbase = blockIdx.y * 64  + wn * 32;
  const int half = lane >> 4;
  const int l15  = lane & 15;

  v8f acc[2][2] = {{(v8f){0.f}, (v8f){0.f}}, {(v8f){0.f}, (v8f){0.f}}};

  const unsigned short* pa_base =
      h + (size_t)(rowbase + l15) * DM + half * 16;
  const int ntg0 = (colbase >> 4);
  const unsigned short* pb_base =
      wp + ((size_t)ntg0 * KSTEPS * 32 + (size_t)lane) * 16;

  for (int kt = 0; kt < KSTEPS; ++kt) {
    Frag32 a[2], b[2];
    const int k0 = kt * 32;
#pragma unroll
    for (int mt = 0; mt < 2; ++mt) {
      const unsigned short* p = pa_base + (size_t)mt * 16 * DM + k0;
      a[mt].u[0] = *(const uint4*)(p);          // K = k0+half*16 .. +7
      a[mt].u[1] = *(const uint4*)(p + 8);      // K = +8 .. +15
    }
#pragma unroll
    for (int nt = 0; nt < 2; ++nt) {
      const unsigned short* p = pb_base + ((size_t)nt * KSTEPS + kt) * 32 * 16;
      b[nt].u[0] = *(const uint4*)(p);
      b[nt].u[1] = *(const uint4*)(p + 8);
    }
    if (kt + 1 < KSTEPS) {   // uniform branch: EXEC stays all-ones
      __builtin_prefetch((const void*)(pa_base + k0 + 32), 0, 0);
      __builtin_prefetch((const void*)(pb_base + (size_t)(kt + 1) * 32 * 16), 0, 0);
    }
#pragma unroll
    for (int mt = 0; mt < 2; ++mt)
#pragma unroll
      for (int nt = 0; nt < 2; ++nt)
        acc[mt][nt] = __builtin_amdgcn_wmma_f32_16x16x32_bf16(
            false, a[mt].v, false, b[nt].v, (short)0, acc[mt][nt], false, false);
  }

  // epilogue: C/D layout -> lane holds col n=l15, rows half*8+r
#pragma unroll
  for (int mt = 0; mt < 2; ++mt) {
    int r0 = rowbase + mt * 16 + half * 8;
#pragma unroll
    for (int nt = 0; nt < 2; ++nt) {
      int col = colbase + nt * 16 + l15;
      float bv = bias[col];
#pragma unroll
      for (int r = 0; r < 8; ++r)
        out[(size_t)(r0 + r) * DM + col] = f2bf(acc[mt][nt][r] + bv);
    }
  }
}

// ---------------- phase 4: per-token 32x32 attn + sequence reduction --------
__global__ __launch_bounds__(256) void gpt_zero_out(float* __restrict__ out) {
  int i = blockIdx.x * 256 + threadIdx.x;
  if (i < NB * HD * HD) out[i] = 0.f;
}

__global__ __launch_bounds__(256) void gpt_attn_reduce(
    const unsigned short* __restrict__ q,
    const unsigned short* __restrict__ k,
    const unsigned short* __restrict__ v,
    float* __restrict__ out) {
  __shared__ float kv[8][32 * 33];      // per-wave staging, reused k then v
  __shared__ float red[DM];
  const int tid = threadIdx.x;
  const int lane = tid & 31, w = tid >> 5;
  const int b = blockIdx.x, chunk = blockIdx.y;

  for (int i = tid; i < DM; i += 256) red[i] = 0.f;
  __syncthreads();

  float accv[32];
#pragma unroll
  for (int l = 0; l < 32; ++l) accv[l] = 0.f;

  union U16 { uint4 u[4]; unsigned short s[32]; };

  for (int t = 0; t < 32; ++t) {
    const int j = chunk * 256 + w * 32 + t;
    const size_t base = ((size_t)b * SEQ + j) * DM;

    U16 qu, ku, vu;
    const uint4* qp = (const uint4*)(q + base + (size_t)lane * 32);
    const uint4* kp = (const uint4*)(k + base + (size_t)lane * 32);
    const uint4* vp = (const uint4*)(v + base + (size_t)lane * 32);
#pragma unroll
    for (int i = 0; i < 4; ++i) { qu.u[i] = qp[i]; ku.u[i] = kp[i]; vu.u[i] = vp[i]; }

    float qrow[32];
#pragma unroll
    for (int l = 0; l < 32; ++l) qrow[l] = bf2f(qu.s[l]);

    // stage k_mat rows (lane == l)
#pragma unroll
    for (int m = 0; m < 32; ++m) kv[w][lane * 33 + m] = bf2f(ku.s[m]);
    __syncthreads();

    // score row (lane == head index): sc[m] = sum_l q[lane][l]*k[l][m]
    float sc[32];
#pragma unroll
    for (int m = 0; m < 32; ++m) sc[m] = 0.f;
#pragma unroll
    for (int l = 0; l < 32; ++l) {
      float ql = qrow[l];
#pragma unroll
      for (int m = 0; m < 32; ++m) sc[m] = fmaf(ql, kv[w][l * 33 + m], sc[m]);
    }
    // softmax over m, scale 1/sqrt(32)
    const float scl = 0.17677669529663687f;
    float mx = -3.0e38f;
#pragma unroll
    for (int m = 0; m < 32; ++m) { sc[m] *= scl; mx = fmaxf(mx, sc[m]); }
    float ssum = 0.f;
#pragma unroll
    for (int m = 0; m < 32; ++m) { sc[m] = __expf(sc[m] - mx); ssum += sc[m]; }
    float inv = 1.f / ssum;
    __syncthreads();

    // stage v_mat rows (lane == m)
#pragma unroll
    for (int l = 0; l < 32; ++l) kv[w][lane * 33 + l] = bf2f(vu.s[l]);
    __syncthreads();

    // out[lane][l] += sum_m p[m]*v[m][l]
#pragma unroll
    for (int m = 0; m < 32; ++m) {
      float pm = sc[m] * inv;
#pragma unroll
      for (int l = 0; l < 32; ++l)
        accv[l] = fmaf(pm, kv[w][m * 33 + l], accv[l]);
    }
    __syncthreads();
  }

  // block reduction: 8 waves share the same (head,l) grid
#pragma unroll
  for (int l = 0; l < 32; ++l) atomicAdd(&red[lane * 32 + l], accv[l]);
  __syncthreads();
  for (int i = tid; i < DM; i += 256) atomicAdd(&out[b * DM + i], red[i]);
}

// ---------------- launch ----------------
extern "C" void kernel_launch(void* const* d_in, const int* in_sizes, int n_in,
                              void* d_out, int out_size, void* d_ws, size_t ws_size,
                              hipStream_t stream) {
  const int*   x   = (const int*)d_in[0];
  const float* emb = (const float*)d_in[1];
  const float* WQ  = (const float*)d_in[2];
  const float* bQ  = (const float*)d_in[3];
  const float* WK  = (const float*)d_in[4];
  const float* bK  = (const float*)d_in[5];
  const float* WV  = (const float*)d_in[6];
  const float* bV  = (const float*)d_in[7];
  float* out = (float*)d_out;

  char* ws = (char*)d_ws;
  const size_t H_BYTES  = (size_t)TOK * DM * 2;   // 32 MB
  const size_t WP_BYTES = (size_t)DM * DM * 2;    // 2 MB each
  unsigned short* h   = (unsigned short*)(ws);
  unsigned short* wpq = (unsigned short*)(ws + H_BYTES);
  unsigned short* wpk = (unsigned short*)(ws + H_BYTES + WP_BYTES);
  unsigned short* wpv = (unsigned short*)(ws + H_BYTES + 2 * WP_BYTES);
  unsigned short* qb  = (unsigned short*)(ws + H_BYTES + 3 * WP_BYTES);
  unsigned short* kb  = (unsigned short*)(ws + 2 * H_BYTES + 3 * WP_BYTES);
  unsigned short* vb  = (unsigned short*)(ws + 3 * H_BYTES + 3 * WP_BYTES);

  gpt_embed_pos<<<TOK, 256, 0, stream>>>(x, emb, h);

  const int packBlocks = (DM * DM) / 256;         // 4096
  gpt_pack_w<<<packBlocks, 256, 0, stream>>>(WQ, wpq);
  gpt_pack_w<<<packBlocks, 256, 0, stream>>>(WK, wpk);
  gpt_pack_w<<<packBlocks, 256, 0, stream>>>(WV, wpv);

  dim3 gg(TOK / 128, DM / 64);                    // 128 x 16
  gpt_gemm_qkv<<<gg, 256, 0, stream>>>(h, wpq, bQ, qb);
  gpt_gemm_qkv<<<gg, 256, 0, stream>>>(h, wpk, bK, kb);
  gpt_gemm_qkv<<<gg, 256, 0, stream>>>(h, wpv, bV, vb);

  gpt_zero_out<<<(NB * HD * HD + 255) / 256, 256, 0, stream>>>(out);
  gpt_attn_reduce<<<dim3(NB, SEQ / 256), 256, 0, stream>>>(qb, kb, vb, out);
}
